// VAE_27960237097244
// MI455X (gfx1250) — compile-verified, matrix operand order checked by reading, not run on hardware
//
#include <hip/hip_runtime.h>
#include <hip/hip_bf16.h>
#include <cmath>

// ---------------- problem constants (match reference) ----------------
constexpr int Bn  = 1024;   // batch
constexpr int DCn = 4;      // data channels
constexpr int Hn  = 64;     // hidden
constexpr int Pn  = 4;      // latent params
constexpr int PLn = 2;      // hidden layers in decoder
constexpr int Ln  = 1024;   // input length
constexpr int Tn  = 512;    // predict length

// encoder lengths after each periodic-pad + dilated VALID conv
constexpr int LO1 = 998;    // 1024+2 - 29 + 1
constexpr int LO2 = 946;    // 998+4  - 57 + 1
constexpr int LO3 = 906;    // 946+8  - 49 + 1
constexpr int LENC= 826;    // 906+16 - 97 + 1

typedef __attribute__((ext_vector_type(16))) __bf16 v16bf;
typedef __attribute__((ext_vector_type(8)))  __bf16 v8bf;
typedef __attribute__((ext_vector_type(4)))  __bf16 v4bf;
typedef __attribute__((ext_vector_type(8)))  float  v8f;
typedef __attribute__((ext_vector_type(4)))  float  v4f;
typedef __attribute__((ext_vector_type(4)))  unsigned int uint32x4;
typedef __attribute__((ext_vector_type(4)))  int    int32x4;
typedef __attribute__((ext_vector_type(8)))  int    int32x8;

// ---------------- workspace layout (bytes, all multiples of 256) ------
constexpr size_t SZ_H1C  = (size_t)Bn*4 *LO1 *4;
constexpr size_t SZ_H2C  = (size_t)Bn*16*LO2 *4;
constexpr size_t SZ_H3C  = (size_t)Bn*64*LO3 *4;
constexpr size_t SZ_H4C  = (size_t)Bn*64*LENC*4;
constexpr size_t OFF_H1C = 0;
constexpr size_t OFF_H2C = OFF_H1C + SZ_H1C;
constexpr size_t OFF_H3C = OFF_H2C + SZ_H2C;
constexpr size_t OFF_H4C = OFF_H3C + SZ_H3C;
constexpr size_t OFF_TAIL= OFF_H4C + SZ_H4C;       // conv region dead after pooling
constexpr size_t OFF_Z   = OFF_TAIL;
constexpr size_t OFF_WIN = OFF_Z    + (size_t)Bn*Pn*4;
constexpr size_t OFF_BIN = OFF_WIN  + (size_t)Bn*DCn*Hn*4;
constexpr size_t OFF_WOUT= OFF_BIN  + (size_t)Bn*Hn*4;
constexpr size_t OFF_BOUT= OFF_WOUT + (size_t)Bn*Hn*DCn*4;
constexpr size_t OFF_WP  = OFF_BOUT + (size_t)Bn*DCn*4;
constexpr size_t OFF_BP  = OFF_WP   + (size_t)Bn*PLn*Hn*Hn*4;
constexpr size_t OFF_ABUF= 0;   // bf16 hidden acts alias dead conv region (max 64 MB)

// =====================================================================
// 1) periodic dilated conv + bias + relu; 4 output positions per thread
// =====================================================================
constexpr int CLV = 4;
__global__ void pconv_kernel(const float* __restrict__ in, const float* __restrict__ w,
                             const float* __restrict__ bias, float* __restrict__ out,
                             int IC, int OC, int Lin, int Lout, int KW, int dil, int pad)
{
    int idx = blockIdx.x * blockDim.x + threadIdx.x;
    int nLblk = (Lout + CLV - 1) / CLV;
    int total = Bn * OC * nLblk;
    if (idx >= total) return;
    int lb = (idx % nLblk) * CLV;
    int oc = (idx / nLblk) % OC;
    int b  = idx / (nLblk * OC);

    float acc[CLV];
#pragma unroll
    for (int j = 0; j < CLV; ++j) acc[j] = bias[oc];

    const float* inb = in + (size_t)b * IC * Lin;
    const float* wo  = w  + (size_t)oc * IC * KW;
    for (int c = 0; c < IC; ++c) {
        const float* ic = inb + (size_t)c * Lin;
        const float* wc = wo + c * KW;
#pragma unroll 4
        for (int k = 0; k < KW; ++k) {
            float wv = wc[k];
            int i0 = lb + k * dil - pad;          // periodic wrap (single wrap suffices)
#pragma unroll
            for (int j = 0; j < CLV; ++j) {
                int i = i0 + j;
                if (i < 0) i += Lin; else if (i >= Lin) i -= Lin;
                acc[j] = fmaf(ic[i], wv, acc[j]);
            }
        }
    }
    float* ob = out + ((size_t)b * OC + oc) * Lout;
#pragma unroll
    for (int j = 0; j < CLV; ++j)
        if (lb + j < Lout) ob[lb + j] = fmaxf(acc[j], 0.f);
}

// =====================================================================
// 2) projections + softmax-attention pooling + logsumexp (one WG per b)
// =====================================================================
__global__ __launch_bounds__(64) void proj_pool_kernel(
    const float* __restrict__ h, const float* __restrict__ pw, const float* __restrict__ pb,
    const float* __restrict__ lw, const float* __restrict__ lb,
    float* __restrict__ zout, float* __restrict__ params_out, float* __restrict__ logvar_out,
    int Lenc, float logconst)
{
    __shared__ float pwS[Pn*Hn], lwS[Pn*Hn];
    __shared__ float mS[Pn][64], sS[Pn][64], psS[Pn][64];
    int b = blockIdx.x, tid = threadIdx.x;
    for (int i = tid; i < Pn*Hn; i += 64) { pwS[i] = pw[i]; lwS[i] = lw[i]; }
    __syncthreads();

    float m[Pn], s[Pn], ps[Pn];
#pragma unroll
    for (int p = 0; p < Pn; ++p) { m[p] = -1e30f; s[p] = 0.f; ps[p] = 0.f; }

    const float* hb = h + (size_t)b * Hn * Lenc;
    for (int l = tid; l < Lenc; l += 64) {
        float hc[Hn];
#pragma unroll
        for (int c = 0; c < Hn; ++c) hc[c] = hb[(size_t)c * Lenc + l];
#pragma unroll
        for (int p = 0; p < Pn; ++p) {
            float pr = pb[p], lv = lb[p];
#pragma unroll
            for (int c = 0; c < Hn; ++c) {
                pr = fmaf(hc[c], pwS[p*Hn+c], pr);
                lv = fmaf(hc[c], lwS[p*Hn+c], lv);
            }
            float a = -lv;                        // streaming logsumexp of -logvar
            if (a > m[p]) {
                float sc = __expf(m[p] - a);
                s[p]  = s[p]  * sc + 1.f;
                ps[p] = ps[p] * sc + pr;
                m[p]  = a;
            } else {
                float e = __expf(a - m[p]);
                s[p]  += e;
                ps[p] = fmaf(e, pr, ps[p]);
            }
        }
    }
#pragma unroll
    for (int p = 0; p < Pn; ++p) { mS[p][tid] = m[p]; sS[p][tid] = s[p]; psS[p][tid] = ps[p]; }
    __syncthreads();
    for (int off = 32; off > 0; off >>= 1) {
        if (tid < off) {
#pragma unroll
            for (int p = 0; p < Pn; ++p) {
                float m1 = mS[p][tid], m2 = mS[p][tid+off];
                float s1 = sS[p][tid], s2 = sS[p][tid+off];
                float p1 = psS[p][tid], p2 = psS[p][tid+off];
                if (m2 > m1) { float t; t=m1;m1=m2;m2=t; t=s1;s1=s2;s2=t; t=p1;p1=p2;p2=t; }
                float sc = __expf(m2 - m1);
                mS[p][tid] = m1; sS[p][tid] = s1 + s2*sc; psS[p][tid] = p1 + p2*sc;
            }
        }
        __syncthreads();
    }
    if (tid < Pn) {
        int p = tid;
        float prm = psS[p][0] / sS[p][0];
        float lvo = -(mS[p][0] + __logf(sS[p][0])) + logconst;
        zout[b*Pn + p]       = prm;
        params_out[b*Pn + p] = prm;
        logvar_out[b*Pn + p] = lvo;
    }
}

// =====================================================================
// 3) hypernet layer 1:  h1 = relu(z @ W1 + b1)  -> bf16
// =====================================================================
__global__ void hyper_l1_kernel(const float* __restrict__ z, const float* __restrict__ w1,
                                const float* __restrict__ b1, __bf16* __restrict__ out, int N1)
{
    int idx = blockIdx.x * blockDim.x + threadIdx.x;
    if (idx >= Bn * N1) return;
    int j = idx % N1, b = idx / N1;
    float acc = b1[j];
    const float* zb = z + b * Pn;
#pragma unroll
    for (int p = 0; p < Pn; ++p) acc = fmaf(zb[p], w1[p*N1 + j], acc);
    out[idx] = (__bf16)fmaxf(acc, 0.f);
}

// =====================================================================
// 3b) tiny fused hypernet (for K1 not multiple of 32): full fp32
// =====================================================================
__global__ void hyper_small_kernel(const float* __restrict__ z,
                                   const float* __restrict__ w1, const float* __restrict__ b1,
                                   const float* __restrict__ w2, const float* __restrict__ b2,
                                   float* __restrict__ out, int N1, int N2)
{
    int idx = blockIdx.x * blockDim.x + threadIdx.x;
    if (idx >= Bn * N2) return;
    int n = idx % N2, b = idx / N2;
    const float* zb = z + b * Pn;
    float acc = b2[n];
    for (int j = 0; j < N1; ++j) {
        float hv = b1[j];
#pragma unroll
        for (int p = 0; p < Pn; ++p) hv = fmaf(zb[p], w1[p*N1 + j], hv);
        acc = fmaf(fmaxf(hv, 0.f), w2[j*N2 + n], acc);
    }
    out[idx] = acc;
}

// =====================================================================
// 4) hypernet layer 2: C[M,N] = A_bf16[M,K] @ W_f32[K,N] + bias  (WMMA)
//    REQUIRES: M % 128 == 0, K % 32 == 0, N % 16 == 0.
//    A tile loaded by the Tensor Data Mover (pad_enable -> 40-half pitch),
//    overlapped with VALU fp32->bf16 staging of the W tile; TENSORcnt is
//    drained only after B staging, just before the barrier.
// =====================================================================
constexpr int GBM = 128, GBN = 128, GBK = 32;
constexpr int APITCH = 40, BPITCH = 40;   // padded row pitch (bf16 units)

__global__ __launch_bounds__(256) void gemm_wmma_kernel(
    const __bf16* __restrict__ A, const float* __restrict__ W,
    const float* __restrict__ bias, float* __restrict__ C,
    int M, int N, int K)
{
    __shared__ __align__(16) __bf16 As[GBM * APITCH];   // [m][k], TDM-padded rows
    __shared__ __align__(16) __bf16 Bs[GBN * BPITCH];   // transposed [n][k]

    int tid  = threadIdx.x;
    int lane = tid & 31, wave = tid >> 5;
    int wm = wave >> 2, wn = wave & 3;                  // 2x4 wave grid
    int tileM = blockIdx.x * GBM;                       // x = M so WGs sharing an
    int tileN = blockIdx.y * GBN;                       // N-slab of W are adjacent (L2 reuse)
    int rowL = lane & 15, hi = lane >> 4;
    int ks0  = hi ? 8 : 0, ks1 = hi ? 24 : 16;          // ISA 16-bit A/B K-halves

    v8f c[4][2] = {};

    for (int kk = 0; kk < K; kk += GBK) {
        // ---- issue A-tile DMA (wave 0); runs while we stage the W tile ----
#if defined(__gfx1250__)
        if (tid < 32) {
            unsigned ldsAddr = (unsigned)(size_t)(&As[0]);
            unsigned long long ga = (unsigned long long)(size_t)(A + (size_t)tileM * K + kk);
            uint32x4 g0;
            g0[0] = 1u;                                             // count=1 valid user D#
            g0[1] = ldsAddr;                                        // lds_addr
            g0[2] = (unsigned)(ga & 0xFFFFFFFFull);                 // global_addr lo
            g0[3] = (unsigned)((ga >> 32) & 0x1FFFFFFull) | (2u << 30); // addr hi | type=2
            int32x8 g1;
            g1[0] = (int)((1u << 16)      // data_size = 2 bytes
                        | (1u << 20)      // pad_enable
                        | (3u << 22)      // pad_interval: 16 DWORDs (=32 bf16 row)
                        | (3u << 25));    // pad_amount: 4 DWORDs (=8 bf16 pad)
            g1[1] = (int)(((unsigned)K & 0xFFFFu) << 16);           // tensor_dim0 lo16
            g1[2] = (int)(((unsigned)K >> 16) | (((unsigned)M & 0xFFFFu) << 16)); // dim0 hi | dim1 lo
            g1[3] = (int)(((unsigned)M >> 16) | (32u << 16));       // dim1 hi | tile_dim0=32
            g1[4] = (int)128u;                                      // tile_dim1=128, tile_dim2=0
            g1[5] = (int)K;                                         // tensor_dim0_stride lo32
            g1[6] = 0;                                              // stride hi | dim1_stride lo
            g1[7] = 0;
            int32x4 gz = {0, 0, 0, 0};
            int32x8 gz8 = {0, 0, 0, 0, 0, 0, 0, 0};
            __builtin_amdgcn_tensor_load_to_lds(g0, g1, gz, gz, gz8, 0);
        }
#endif
        // ---- stage W tile fp32 -> bf16, transposed into Bs[n][k] (overlaps TDM) ----
#pragma unroll
        for (int i = 0; i < 4; ++i) {
            int v  = tid + 256*i;        // 0..1023
            int n  = v >> 3;             // 0..127
            int k0 = (v & 7) * 4;        // 0..28
            int gn = tileN + n;
            v4bf bv;
            if (gn < N) {
                const float* src = &W[(size_t)(kk + k0) * N + gn];
                bv[0] = (__bf16)src[0];
                bv[1] = (__bf16)src[(size_t)N];
                bv[2] = (__bf16)src[(size_t)2*N];
                bv[3] = (__bf16)src[(size_t)3*N];
            } else {
                bv[0] = bv[1] = bv[2] = bv[3] = (__bf16)0.f;
            }
            *(v4bf*)&Bs[n*BPITCH + k0] = bv;
        }

        // ---- prefetch next W tile (global_prefetch_b8) ----
        if (kk + GBK < K) {
            int k = tid >> 3, n = (tid & 7) * 16;
            if (tileN + n < N)
                __builtin_prefetch(&W[(size_t)(kk + GBK + k)*N + tileN + n], 0, 1);
        }

        // ---- drain the A-tile DMA only now, then publish both tiles ----
#if defined(__gfx1250__)
        if (tid < 32) __builtin_amdgcn_s_wait_tensorcnt(0);
#endif
        __syncthreads();

        // ---- load fragments per ISA 16-bit layouts ----
        v16bf a[4], bf[2];
#pragma unroll
        for (int fm = 0; fm < 4; ++fm) {
            const __bf16* base = &As[(wm*64 + fm*16 + rowL) * APITCH];
            v8bf lo = *(const v8bf*)(base + ks0);
            v8bf hi8= *(const v8bf*)(base + ks1);
            a[fm] = __builtin_shufflevector(lo, hi8, 0,1,2,3,4,5,6,7,8,9,10,11,12,13,14,15);
        }
#pragma unroll
        for (int fn = 0; fn < 2; ++fn) {
            const __bf16* base = &Bs[(wn*32 + fn*16 + rowL) * BPITCH];
            v8bf lo = *(const v8bf*)(base + ks0);
            v8bf hi8= *(const v8bf*)(base + ks1);
            bf[fn] = __builtin_shufflevector(lo, hi8, 0,1,2,3,4,5,6,7,8,9,10,11,12,13,14,15);
        }
#pragma unroll
        for (int fm = 0; fm < 4; ++fm)
#pragma unroll
            for (int fn = 0; fn < 2; ++fn)
                c[fm][fn] = __builtin_amdgcn_wmma_f32_16x16x32_bf16(
                    false, a[fm], false, bf[fn], (short)0, c[fm][fn], false, false);
        __syncthreads();   // protects As/Bs against next iteration's writes
    }

    // ---- store: lane<16 -> M=r, lane>=16 -> M=r+8 ; N = lane&15 ----
#pragma unroll
    for (int fm = 0; fm < 4; ++fm)
#pragma unroll
        for (int fn = 0; fn < 2; ++fn) {
            int n_g = tileN + wn*32 + fn*16 + rowL;
            if (n_g >= N) continue;
            float bv = bias[n_g];
            int m_g = tileM + wm*64 + fm*16 + hi*8;     // M % 128 == 0: no M guard
#pragma unroll
            for (int r = 0; r < 8; ++r)
                C[(size_t)(m_g + r)*N + n_g] = c[fm][fn][r] + bv;
        }
}

// =====================================================================
// 5) recurrent rollout: one 64-thread WG per sample, weights in LDS
// =====================================================================
__global__ __launch_bounds__(64) void scan_kernel(
    const float* __restrict__ y0,  const float* __restrict__ Win,
    const float* __restrict__ bin, const float* __restrict__ Wout,
    const float* __restrict__ bout,const float* __restrict__ Wp,
    const float* __restrict__ bp,  const float* __restrict__ cutoff,
    float* __restrict__ yout)
{
    extern __shared__ float lds[];
    float* wpS   = lds;                    // PLn*Hn*Hn = 8192
    float* winS  = wpS   + PLn*Hn*Hn;      // 256 ([c][h])
    float* woutS = winS  + DCn*Hn;         // 256 ([h][c])
    float* binS  = woutS + Hn*DCn;         // 64
    float* bpS   = binS  + Hn;             // 128
    float* boutS = bpS   + PLn*Hn;         // 4
    float* uS    = boutS + DCn;            // 64
    float* u2S   = uS    + Hn;             // 64
    float* yS    = u2S   + Hn;             // 4

    int b = blockIdx.x, tid = threadIdx.x;
    for (int i = tid; i < PLn*Hn*Hn; i += 64) wpS[i] = Wp[(size_t)b*PLn*Hn*Hn + i];
    for (int i = tid; i < DCn*Hn;   i += 64) { winS[i] = Win[b*DCn*Hn + i]; woutS[i] = Wout[b*Hn*DCn + i]; }
    binS[tid] = bin[b*Hn + tid];
    for (int i = tid; i < PLn*Hn; i += 64) bpS[i] = bp[b*PLn*Hn + i];
    if (tid < DCn) { boutS[tid] = bout[b*DCn + tid]; yS[tid] = y0[b*DCn + tid]; }
    float cut = cutoff[0];
    const float dt = 1e-6f;
    __syncthreads();

    for (int t = 0; t < Tn; ++t) {
        { // input layer: u = relu(W_in^T y + b_in)
            float acc = binS[tid];
#pragma unroll
            for (int c = 0; c < DCn; ++c) acc = fmaf(yS[c], winS[c*Hn + tid], acc);
            uS[tid] = fmaxf(acc, 0.f);
        }
        __syncthreads();
        { // hidden layer 0
            float acc = bpS[tid];
#pragma unroll 8
            for (int h = 0; h < Hn; ++h) acc = fmaf(uS[h], wpS[h*Hn + tid], acc);
            u2S[tid] = fmaxf(acc, 0.f);
        }
        __syncthreads();
        { // hidden layer 1
            float acc = bpS[Hn + tid];
#pragma unroll 8
            for (int h = 0; h < Hn; ++h) acc = fmaf(u2S[h], wpS[Hn*Hn + h*Hn + tid], acc);
            uS[tid] = fmaxf(acc, 0.f);
        }
        __syncthreads();
        if (tid < DCn) { // output layer + state update + store
            float acc = boutS[tid];
#pragma unroll 8
            for (int h = 0; h < Hn; ++h) acc = fmaf(uS[h], woutS[h*DCn + tid], acc);
            float yv = yS[tid] + cut * tanhf(dt * acc / cut);
            yS[tid] = yv;
            yout[((size_t)b*DCn + tid)*Tn + t] = yv;
        }
        __syncthreads();
    }
}

// =====================================================================
// host side
// =====================================================================
static void run_hyper(const float* z, const float* w1, const float* b1,
                      const float* w2, const float* b2, int N1, int N2,
                      __bf16* Abuf, float* Cout, hipStream_t stream)
{
    int tot = Bn * N1;
    hyper_l1_kernel<<<(tot + 255)/256, 256, 0, stream>>>(z, w1, b1, Abuf, N1);
    dim3 g((Bn + GBM - 1)/GBM, (N2 + GBN - 1)/GBN);
    gemm_wmma_kernel<<<g, 256, 0, stream>>>(Abuf, w2, b2, Cout, Bn, N2, N1);
}

extern "C" void kernel_launch(void* const* d_in, const int* in_sizes, int n_in,
                              void* d_out, int out_size, void* d_ws, size_t ws_size,
                              hipStream_t stream)
{
    (void)in_sizes; (void)n_in; (void)out_size; (void)ws_size;
    const float* x    = (const float*)d_in[0];
    const float* y0   = (const float*)d_in[1];
    const float* c1w  = (const float*)d_in[2];  const float* c1b  = (const float*)d_in[3];
    const float* c2w  = (const float*)d_in[4];  const float* c2b  = (const float*)d_in[5];
    const float* c3w  = (const float*)d_in[6];  const float* c3b  = (const float*)d_in[7];
    const float* c4w  = (const float*)d_in[8];  const float* c4b  = (const float*)d_in[9];
    const float* pw   = (const float*)d_in[10]; const float* pb   = (const float*)d_in[11];
    const float* lw   = (const float*)d_in[12]; const float* lb   = (const float*)d_in[13];
    const float* inw1 = (const float*)d_in[14]; const float* inb1 = (const float*)d_in[15];
    const float* inw2 = (const float*)d_in[16]; const float* inb2 = (const float*)d_in[17];
    const float* ibw1 = (const float*)d_in[18]; const float* ibb1 = (const float*)d_in[19];
    const float* ibw2 = (const float*)d_in[20]; const float* ibb2 = (const float*)d_in[21];
    const float* ow1  = (const float*)d_in[22]; const float* ob1  = (const float*)d_in[23];
    const float* ow2  = (const float*)d_in[24]; const float* ob2  = (const float*)d_in[25];
    const float* obw1 = (const float*)d_in[26]; const float* obb1 = (const float*)d_in[27];
    const float* obw2 = (const float*)d_in[28]; const float* obb2 = (const float*)d_in[29];
    const float* pww1 = (const float*)d_in[30]; const float* pwb1 = (const float*)d_in[31];
    const float* pww2 = (const float*)d_in[32]; const float* pwb2 = (const float*)d_in[33];
    const float* pbw1 = (const float*)d_in[34]; const float* pbb1 = (const float*)d_in[35];
    const float* pbw2 = (const float*)d_in[36]; const float* pbb2 = (const float*)d_in[37];
    const float* cutf = (const float*)d_in[38];

    char* ws = (char*)d_ws;
    float* h1c  = (float*)(ws + OFF_H1C);
    float* h2c  = (float*)(ws + OFF_H2C);
    float* h3c  = (float*)(ws + OFF_H3C);
    float* h4c  = (float*)(ws + OFF_H4C);
    float* zbuf = (float*)(ws + OFF_Z);
    float* WinB = (float*)(ws + OFF_WIN);
    float* binB = (float*)(ws + OFF_BIN);
    float* WoutB= (float*)(ws + OFF_WOUT);
    float* boutB= (float*)(ws + OFF_BOUT);
    float* WpB  = (float*)(ws + OFF_WP);
    float* bpB  = (float*)(ws + OFF_BP);
    __bf16* Abuf= (__bf16*)(ws + OFF_ABUF);     // aliases conv region (dead by then)

    float* yout       = (float*)d_out;
    float* params_out = yout + (size_t)Bn*DCn*Tn;
    float* logvar_out = params_out + (size_t)Bn*Pn;

    // ---- encoder ----
    auto blks = [](long long n){ return (unsigned)((n + 255) / 256); };
    auto lb4  = [](int L){ return (long long)((L + CLV - 1) / CLV); };
    pconv_kernel<<<blks(Bn*4LL *lb4(LO1)), 256, 0, stream>>>(x,   c1w, c1b, h1c,  DCn, 4,  Ln,  LO1, 8, 4, 1);
    pconv_kernel<<<blks(Bn*16LL*lb4(LO2)), 256, 0, stream>>>(h1c, c2w, c2b, h2c,  4,  16, LO1, LO2, 8, 8, 2);
    pconv_kernel<<<blks(Bn*64LL*lb4(LO3)), 256, 0, stream>>>(h2c, c3w, c3b, h3c,  16, 64, LO2, LO3, 4, 16, 4);
    pconv_kernel<<<blks(Bn*64LL*lb4(LENC)),256, 0, stream>>>(h3c, c4w, c4b, h4c,  64, 64, LO3, LENC,4, 32, 8);

    // ---- pooling:  const = max(1, 0.9*min(31,P)*min(31,Lenc)) ----
    float cst = 0.9f * (float)((Pn < 31) ? Pn : 31) * (float)((LENC < 31) ? LENC : 31);
    if (cst < 1.f) cst = 1.f;
    proj_pool_kernel<<<Bn, 64, 0, stream>>>(h4c, pw, pb, lw, lb, zbuf,
                                            params_out, logvar_out, LENC, logf(cst));

    // ---- hypernetworks (layer1 VALU, layer2 WMMA bf16; tiny one fused) ----
    run_hyper(zbuf, inw1, inb1, inw2, inb2, 4*DCn*Hn,    DCn*Hn,    Abuf, WinB,  stream); // W_in  (B,4,64)
    run_hyper(zbuf, ibw1, ibb1, ibw2, ibb2, 4*Hn,        Hn,        Abuf, binB,  stream); // b_in  (B,64)
    run_hyper(zbuf, ow1,  ob1,  ow2,  ob2,  4*DCn*Hn,    DCn*Hn,    Abuf, WoutB, stream); // W_out (B,64,4)
    hyper_small_kernel<<<blks((long long)Bn*DCn), 256, 0, stream>>>(                      // b_out (B,4): K=16
        zbuf, obw1, obb1, obw2, obb2, boutB, 4*DCn, DCn);
    run_hyper(zbuf, pww1, pwb1, pww2, pwb2, 4*PLn*Hn*Hn, PLn*Hn*Hn, Abuf, WpB,   stream); // Wp    (B,2,64,64)
    run_hyper(zbuf, pbw1, pbb1, pbw2, pbb2, 4*PLn*Hn,    PLn*Hn,    Abuf, bpB,   stream); // bp    (B,2,64)

    // ---- recurrent rollout ----
    size_t scan_lds = (size_t)(PLn*Hn*Hn + DCn*Hn + Hn*DCn + Hn + PLn*Hn + DCn + Hn + Hn + DCn) * sizeof(float);
    scan_kernel<<<Bn, 64, scan_lds, stream>>>(y0, WinB, binB, WoutB, boutB, WpB, bpB, cutf, yout);
}
